// GNN_61340722921704
// MI455X (gfx1250) — matrix operand order repaired and checked by reading, not compiled, not run on hardware
//
#include <hip/hip_runtime.h>
#include <hip/hip_bf16.h>

#define NFEAT 128
#define NHEAD 4

typedef __attribute__((ext_vector_type(16))) _Float16 v16h;
typedef __attribute__((ext_vector_type(8)))  float    v8f;

__device__ __forceinline__ float leakyrelu(float v) { return v > 0.f ? v : 0.2f * v; }

// float atomic max via int/uint trick (valid for all finite values + -inf init)
__device__ __forceinline__ void atomicMaxF(float* addr, float val) {
  if (val >= 0.f) atomicMax((int*)addr, __float_as_int(val));
  else            atomicMin((unsigned int*)addr, __float_as_uint(val));
}

// ---------------------------------------------------------------------------
// Zero / init: agg[N*128]=0, m[N*H]=-inf, s[N*H]=0
// ---------------------------------------------------------------------------
__global__ void k_init(float* __restrict__ agg, float* __restrict__ m,
                       float* __restrict__ s, int N) {
  int tid = blockIdx.x * blockDim.x + threadIdx.x;
  if (tid < N * NFEAT) agg[tid] = 0.f;
  if (tid < N * NHEAD) { m[tid] = -__builtin_inff(); s[tid] = 0.f; }
}

// ---------------------------------------------------------------------------
// Swizzle fp32 row-major [M,128] -> f16 WMMA A-fragment layout
// index = ((mt*4 + kb)*32 + lane)*16 + i ; per ISA 16-bit A 16x32 layout:
//   row = mt*16 + lane%16 ; v=i/2 ; k = kb*32 + (v>=4)*16 + (lane/16)*8 + (v%4)*2 + i%2
// ---------------------------------------------------------------------------
__global__ void k_swizzleA(const float* __restrict__ x, _Float16* __restrict__ xh,
                           int total, int M) {
  int tid = blockIdx.x * blockDim.x + threadIdx.x;
  if (tid >= total) return;
  int i    = tid & 15;
  int lane = (tid >> 4) & 31;
  int kb   = (tid >> 9) & 3;
  int mt   = tid >> 11;
  int row  = mt * 16 + (lane & 15);
  int v    = i >> 1;
  int k    = kb * 32 + ((v >> 2) << 4) + ((lane >> 4) << 3) + ((v & 3) << 1) + (i & 1);
  float val = (row < M) ? x[(size_t)row * NFEAT + k] : 0.f;
  xh[tid] = (_Float16)val;
}

// Same, fused with bias + ReLU (layer-2 input = relu(agg + b))
__global__ void k_bias_relu_swizzleA(const float* __restrict__ agg,
                                     const float* __restrict__ b,
                                     _Float16* __restrict__ xh, int total, int M) {
  int tid = blockIdx.x * blockDim.x + threadIdx.x;
  if (tid >= total) return;
  int i    = tid & 15;
  int lane = (tid >> 4) & 31;
  int kb   = (tid >> 9) & 3;
  int mt   = tid >> 11;
  int row  = mt * 16 + (lane & 15);
  int v    = i >> 1;
  int k    = kb * 32 + ((v >> 2) << 4) + ((lane >> 4) << 3) + ((v & 3) << 1) + (i & 1);
  float val = (row < M) ? fmaxf(agg[(size_t)row * NFEAT + k] + b[k], 0.f) : 0.f;
  xh[tid] = (_Float16)val;
}

// ---------------------------------------------------------------------------
// Swizzle W [128,128] fp32 row-major (K,N) -> f16 WMMA B-fragment layout
// index = ((nt*4 + kb)*32 + lane)*16 + i ; col = nt*16 + lane%16 ;
// k = kb*32 + (i/8)*16 + (lane/16)*8 + i%8   (same k order as the A fragment)
// ---------------------------------------------------------------------------
__global__ void k_swizzleB(const float* __restrict__ W, _Float16* __restrict__ Wh) {
  int tid = blockIdx.x * blockDim.x + threadIdx.x;
  if (tid >= 8 * 4 * 32 * 16) return;
  int i    = tid & 15;
  int lane = (tid >> 4) & 31;
  int kb   = (tid >> 9) & 3;
  int nt   = tid >> 11;
  int col  = nt * 16 + (lane & 15);
  int k    = kb * 32 + ((i >> 3) << 4) + ((lane >> 4) << 3) + (i & 7);
  Wh[tid] = (_Float16)W[(size_t)k * NFEAT + col];
}

// ---------------------------------------------------------------------------
// GEMM: C[mtiles*16,128] = A x B. One wave per 16-row tile, all 8 N-tiles per
// wave; A fragments stay resident; 32 v_wmma per wave. C is padded to a
// whole number of tiles, so the store epilogue is fully unguarded: one
// 8-store clause with immediate offsets per N-tile.
// ---------------------------------------------------------------------------
__global__ void k_gemm(const _Float16* __restrict__ A, const _Float16* __restrict__ B,
                       float* __restrict__ C, int mtiles) {
  int gwave = (int)((blockIdx.x * blockDim.x + threadIdx.x) >> 5);
  int lane  = threadIdx.x & 31;
  if (gwave >= mtiles) return;
  const v16h* Ap = ((const v16h*)A) + (size_t)gwave * 128 + lane; // 4 kb * 32 lanes
  const v16h* Bp = ((const v16h*)B) + lane;
  v16h a0 = Ap[0], a1 = Ap[32], a2 = Ap[64], a3 = Ap[96];
  int rbase = gwave * 16 + ((lane >> 4) << 3);
  float* Crow = C + (size_t)rbase * NFEAT + (lane & 15);
#pragma unroll
  for (int nt = 0; nt < 8; ++nt) {
    v8f acc = {};
    acc = __builtin_amdgcn_wmma_f32_16x16x32_f16(false, a0, false, Bp[(nt * 4 + 0) * 32],
                                                 (short)0, acc, false, false);
    acc = __builtin_amdgcn_wmma_f32_16x16x32_f16(false, a1, false, Bp[(nt * 4 + 1) * 32],
                                                 (short)0, acc, false, false);
    acc = __builtin_amdgcn_wmma_f32_16x16x32_f16(false, a2, false, Bp[(nt * 4 + 2) * 32],
                                                 (short)0, acc, false, false);
    acc = __builtin_amdgcn_wmma_f32_16x16x32_f16(false, a3, false, Bp[(nt * 4 + 3) * 32],
                                                 (short)0, acc, false, false);
    float* Cp = Crow + nt * 16;
#pragma unroll
    for (int r = 0; r < 8; ++r) Cp[(size_t)r * NFEAT] = acc[r];
  }
}

// ---------------------------------------------------------------------------
// Per-node attention logits: al[n,h] = dot(xp[n,h,:], a[h,:])
// ---------------------------------------------------------------------------
__global__ void k_logits(const float* __restrict__ xp, const float* __restrict__ a_src,
                         const float* __restrict__ a_dst, float* __restrict__ alS,
                         float* __restrict__ alD, int N) {
  int tid = blockIdx.x * blockDim.x + threadIdx.x;
  if (tid >= N * NHEAD) return;
  int n = tid >> 2, h = tid & 3;
  const float* xr = xp + (size_t)n * NFEAT + h * 32;
  const float* as = a_src + h * 32;
  const float* ad = a_dst + h * 32;
  float s1 = 0.f, s2 = 0.f;
#pragma unroll
  for (int c = 0; c < 32; ++c) { float v = xr[c]; s1 += v * as[c]; s2 += v * ad[c]; }
  alS[tid] = s1;
  alD[tid] = s2;
}

// e = leaky_relu(alS[src] + alD[dst]); store; segment-max into m[dst]
__global__ void k_edge_max(const int* __restrict__ src, const int* __restrict__ dst,
                           const float* __restrict__ alS, const float* __restrict__ alD,
                           float* __restrict__ ews, float* __restrict__ m, int E) {
  int tid = blockIdx.x * blockDim.x + threadIdx.x;
  if (tid >= E * NHEAD) return;
  int e = tid >> 2, h = tid & 3;
  int sN = src[e], dN = dst[e];
  float v = leakyrelu(alS[sN * NHEAD + h] + alD[dN * NHEAD + h]);
  ews[tid] = v;
  atomicMaxF(&m[dN * NHEAD + h], v);
}

// p = exp(e - m[dst]); store; segment-sum into s[dst]
__global__ void k_edge_exp(const int* __restrict__ dst, const float* __restrict__ m,
                           float* __restrict__ ews, float* __restrict__ s, int E) {
  int tid = blockIdx.x * blockDim.x + threadIdx.x;
  if (tid >= E * NHEAD) return;
  int e = tid >> 2, h = tid & 3;
  int dN = dst[e];
  float p = __expf(ews[tid] - m[dN * NHEAD + h]);
  ews[tid] = p;
  atomicAdd(&s[dN * NHEAD + h], p);
}

// agg[dst] += xp[src] * alpha ; one thread per (edge, 4 channels), float4 gather
__global__ void k_aggregate(const int* __restrict__ src, const int* __restrict__ dst,
                            const float* __restrict__ xp, const float* __restrict__ p,
                            const float* __restrict__ s, float* __restrict__ agg, int E) {
  int tid = blockIdx.x * blockDim.x + threadIdx.x;
  if (tid >= E * 32) return;
  int e  = tid >> 5;
  int q  = tid & 31;      // 4-channel chunk index, 0..31
  int c4 = q << 2;        // channel base, 0..124
  int h  = q >> 3;        // head = c4/32
  int sN = src[e], dN = dst[e];
  float alpha = p[e * NHEAD + h] / (s[dN * NHEAD + h] + 1e-16f);
  const float4 v = *((const float4*)(xp + (size_t)sN * NFEAT) + q);
  float* o = agg + (size_t)dN * NFEAT + c4;
  atomicAdd(o + 0, v.x * alpha);
  atomicAdd(o + 1, v.y * alpha);
  atomicAdd(o + 2, v.z * alpha);
  atomicAdd(o + 3, v.w * alpha);
}

// out = relu(agg + b), fp32
__global__ void k_bias_relu_out(const float* __restrict__ agg, const float* __restrict__ b,
                                float* __restrict__ out, int N) {
  int tid = blockIdx.x * blockDim.x + threadIdx.x;
  if (tid >= N * NFEAT) return;
  out[tid] = fmaxf(agg[tid] + b[tid & (NFEAT - 1)], 0.f);
}

// ---------------------------------------------------------------------------
extern "C" void kernel_launch(void* const* d_in, const int* in_sizes, int n_in,
                              void* d_out, int out_size, void* d_ws, size_t ws_size,
                              hipStream_t stream) {
  (void)n_in; (void)out_size; (void)ws_size;
  const float* x     = (const float*)d_in[0];
  const int*   src   = (const int*)  d_in[1];
  const int*   dst   = (const int*)  d_in[2];
  const float* W0    = (const float*)d_in[3];
  const float* asrc0 = (const float*)d_in[4];
  const float* adst0 = (const float*)d_in[5];
  const float* b0    = (const float*)d_in[6];
  const float* W1    = (const float*)d_in[7];
  const float* asrc1 = (const float*)d_in[8];
  const float* adst1 = (const float*)d_in[9];
  const float* b1    = (const float*)d_in[10];

  const int N = in_sizes[0] / NFEAT;
  const int E = in_sizes[1];
  const int mtiles = (N + 15) / 16;
  const int Mpad   = mtiles * 16;            // padded row count for GEMM output

  char* base = (char*)d_ws;
  size_t off = 0;
  auto carve = [&](size_t bytes) -> char* {
    char* p = base + off;
    off += (bytes + 255) & ~(size_t)255;
    return p;
  };
  _Float16* xh  = (_Float16*)carve((size_t)mtiles * 2048 * sizeof(_Float16));
  _Float16* Wh  = (_Float16*)carve((size_t)16384 * sizeof(_Float16));
  float*    xp  = (float*)carve((size_t)Mpad * NFEAT * sizeof(float)); // padded
  float*    agg = (float*)carve((size_t)N * NFEAT * sizeof(float));
  float*    alS = (float*)carve((size_t)N * NHEAD * sizeof(float));
  float*    alD = (float*)carve((size_t)N * NHEAD * sizeof(float));
  float*    mws = (float*)carve((size_t)N * NHEAD * sizeof(float));
  float*    sws = (float*)carve((size_t)N * NHEAD * sizeof(float));
  float*    ews = (float*)carve((size_t)E * NHEAD * sizeof(float));

  const int BS = 256;
  const int totalA  = mtiles * 2048;
  const int gInit   = (N * NFEAT + BS - 1) / BS;
  const int gSwzA   = (totalA + BS - 1) / BS;
  const int gSwzB   = (16384 + BS - 1) / BS;
  const int gGemm   = (mtiles + 7) / 8;      // 8 waves / 256-thread block
  const int gLogit  = (N * NHEAD + BS - 1) / BS;
  const int gEdge   = (E * NHEAD + BS - 1) / BS;
  const int gAggr   = (E * 32 + BS - 1) / BS;

  // ----- Layer 1 -----
  k_init<<<gInit, BS, 0, stream>>>(agg, mws, sws, N);
  k_swizzleA<<<gSwzA, BS, 0, stream>>>(x, xh, totalA, N);
  k_swizzleB<<<gSwzB, BS, 0, stream>>>(W0, Wh);
  k_gemm<<<gGemm, BS, 0, stream>>>(xh, Wh, xp, mtiles);
  k_logits<<<gLogit, BS, 0, stream>>>(xp, asrc0, adst0, alS, alD, N);
  k_edge_max<<<gEdge, BS, 0, stream>>>(src, dst, alS, alD, ews, mws, E);
  k_edge_exp<<<gEdge, BS, 0, stream>>>(dst, mws, ews, sws, E);
  k_aggregate<<<gAggr, BS, 0, stream>>>(src, dst, xp, ews, sws, agg, E);

  // ----- Layer 2 -----
  k_bias_relu_swizzleA<<<gSwzA, BS, 0, stream>>>(agg, b0, xh, totalA, N);
  k_init<<<gInit, BS, 0, stream>>>(agg, mws, sws, N);   // after agg consumed
  k_swizzleB<<<gSwzB, BS, 0, stream>>>(W1, Wh);
  k_gemm<<<gGemm, BS, 0, stream>>>(xh, Wh, xp, mtiles);
  k_logits<<<gLogit, BS, 0, stream>>>(xp, asrc1, adst1, alS, alD, N);
  k_edge_max<<<gEdge, BS, 0, stream>>>(src, dst, alS, alD, ews, mws, E);
  k_edge_exp<<<gEdge, BS, 0, stream>>>(dst, mws, ews, sws, E);
  k_aggregate<<<gAggr, BS, 0, stream>>>(src, dst, xp, ews, sws, agg, E);
  k_bias_relu_out<<<gInit, BS, 0, stream>>>(agg, b1, (float*)d_out, N);
}
